// FIBERCrossModalLayer_48292612276896
// MI455X (gfx1250) — matrix operand order repaired and checked by reading, not compiled, not run on hardware
//
#include <hip/hip_runtime.h>
#include <hip/hip_bf16.h>

typedef __bf16 bf16;
typedef __attribute__((ext_vector_type(8)))  __bf16 bf16x8;
typedef __attribute__((ext_vector_type(16))) __bf16 v16bf;
typedef __attribute__((ext_vector_type(8)))  float  v8f;

#define DEV static __device__ __forceinline__

static constexpr int Hd   = 1024;
static constexpr int NH   = 16;
static constexpr int HD   = 64;
static constexpr int Bb   = 8;
static constexpr int LV   = 576;
static constexpr int LT   = 512;

DEV bf16 tobf(float f) { return (bf16)f; }

// scalar lane-XOR butterfly step via ds_swizzle_b32 (used in LayerNorm)
template <int MASK>
DEV float swz(float x) {
  return __int_as_float(__builtin_amdgcn_ds_swizzle(__float_as_int(x),
                                                    0x1f | (MASK << 10)));
}

// 8 back-to-back ds_swizzle_b32 + one wait, forced via one asm bundle so the
// DS pipe stays full (offsets: 0x1f | mask<<10 -> 1055/2079/4127/8223).
#define DEF_SWZ8(NAME, OFFSTR)                                                 \
DEV void NAME(const float* in, float* out) {                                   \
  asm volatile("ds_swizzle_b32 %0, %8 offset:" OFFSTR "\n\t"                   \
               "ds_swizzle_b32 %1, %9 offset:" OFFSTR "\n\t"                   \
               "ds_swizzle_b32 %2, %10 offset:" OFFSTR "\n\t"                  \
               "ds_swizzle_b32 %3, %11 offset:" OFFSTR "\n\t"                  \
               "ds_swizzle_b32 %4, %12 offset:" OFFSTR "\n\t"                  \
               "ds_swizzle_b32 %5, %13 offset:" OFFSTR "\n\t"                  \
               "ds_swizzle_b32 %6, %14 offset:" OFFSTR "\n\t"                  \
               "ds_swizzle_b32 %7, %15 offset:" OFFSTR "\n\t"                  \
               "s_wait_dscnt 0x0"                                              \
               : "=&v"(out[0]), "=&v"(out[1]), "=&v"(out[2]), "=&v"(out[3]),   \
                 "=&v"(out[4]), "=&v"(out[5]), "=&v"(out[6]), "=&v"(out[7])    \
               : "v"(in[0]), "v"(in[1]), "v"(in[2]), "v"(in[3]),               \
                 "v"(in[4]), "v"(in[5]), "v"(in[6]), "v"(in[7]));              \
}
DEF_SWZ8(swz8_x1, "1055")
DEF_SWZ8(swz8_x2, "2079")
DEF_SWZ8(swz8_x4, "4127")
DEF_SWZ8(swz8_x8, "8223")

DEV void maxstep8(float v[8], void (*f)(const float*, float*)) {
  float t[8];
  f(v, t);
#pragma unroll
  for (int r = 0; r < 8; ++r) v[r] = fmaxf(v[r], t[r]);
}

DEV v16bf mkfrag(bf16x8 lo, bf16x8 hi) {
  v16bf f;
#pragma unroll
  for (int i = 0; i < 8; ++i) { f[i] = lo[i]; f[i + 8] = hi[i]; }
  return f;
}

// A-matrix fragment (16x32 bf16): lane m = row; half-wave selects K base 0 or 8;
// elements 0..7 = K kb..kb+7, elements 8..15 = K kb+16..kb+23.
DEV v16bf afrag(const bf16* base, int stride, int m, int kb) {
  bf16x8 lo = *(const bf16x8*)(base + (size_t)m * stride + kb);
  bf16x8 hi = *(const bf16x8*)(base + (size_t)m * stride + kb + 16);
  return mkfrag(lo, hi);
}

// B-matrix fragment (32x16 bf16) with the reduction dim contiguous per column:
// lane n = column; half-wave (kb 0 or 16) selects K base 0 or 8;
// elements 0..7 = K base..base+7, elements 8..15 = K base+16..base+23.
DEV v16bf bfrag(const bf16* base, int stride, int n, int kb) {
  bf16x8 lo = *(const bf16x8*)(base + (size_t)n * stride + kb / 2);
  bf16x8 hi = *(const bf16x8*)(base + (size_t)n * stride + kb / 2 + 16);
  return mkfrag(lo, hi);
}

DEV v8f wmma_bf16(v16bf a, v16bf b, v8f c) {
  return __builtin_amdgcn_wmma_f32_16x16x32_bf16(false, a, false, b, (short)0, c,
                                                 false, false);
}

// ---------------- conversion kernels ----------------

__global__ __launch_bounds__(256) void cvt_bf16_kernel(const float* __restrict__ in,
                                                       bf16* __restrict__ out, int n) {
  int i = blockIdx.x * 256 + threadIdx.x;
  if (i < n) out[i] = tobf(in[i]);
}

// W [K=1024][N=1024] f32 row-major -> Wt [N][K] bf16 (reduction dim contiguous)
__global__ __launch_bounds__(256) void cvt_wT_kernel(const float* __restrict__ W,
                                                     bf16* __restrict__ Wt) {
  int i = blockIdx.x * 256 + threadIdx.x;   // 1M elements
  int k = i & 1023, n = i >> 10;
  Wt[(size_t)n * 1024 + k] = tobf(W[(size_t)k * 1024 + n]);
}

// V [B*L, 1024] bf16 -> Vt [B, NH, HD, L] bf16 (key dim contiguous for PV B-frags)
__global__ __launch_bounds__(256) void vtrans_kernel(const bf16* __restrict__ V,
                                                     bf16* __restrict__ Vt, int L) {
  size_t i = (size_t)blockIdx.x * 256 + threadIdx.x;  // over B*L*1024
  int c = (int)(i & 1023);
  size_t bl = i >> 10;
  int l = (int)(bl % L);
  int b = (int)(bl / L);
  int h = c >> 6, d = c & 63;
  Vt[(((size_t)(b * NH + h)) * HD + d) * L + l] = V[i];
}

// ---------------- GEMM: out[M,1024] = A[M,1024](bf16) @ Wt^T + bias ----------------
// 256 threads = 8 waves; block tile 128x64; K-step 64 per LDS stage; double-buffered
// LDS (48KB) so global_load_b128s for stage k+1 are in flight under stage-k WMMAs;
// global_prefetch for stage k+2. 8 WMMAs per barrier. Branch-free templated epilogue.

template <bool F32OUT>
__global__ __launch_bounds__(256) void gemm_bf16_kernel(
    const bf16* __restrict__ A, const bf16* __restrict__ Wt,
    const float* __restrict__ bias, float* __restrict__ outF,
    bf16* __restrict__ outB, int M) {
  __shared__ bf16 sA[2][128 * 64];
  __shared__ bf16 sB[2][64 * 64];

  const int t = threadIdx.x;
  const int lane = t & 31;
  const int wave = t >> 5;
  const int waveM = wave & 3;       // 4 waves along M (32 rows each)
  const int waveN = wave >> 2;      // 2 waves along N (32 cols each)
  const int m0 = blockIdx.x * 128;
  const int n0 = blockIdx.y * 64;

  const int mloc = lane & 15;
  const int kbA = (lane < 16) ? 0 : 8;
  const int kbB = (lane < 16) ? 0 : 16;

  // staging assignments: A tile 128x64 (32 elems/thread), B tile 64x64 (16/thread)
  const int arow = t >> 1, aseg = (t & 1) * 32;
  const int bcol = t >> 2, bseg = (t & 3) * 16;
  const bf16* aptr = A  + (size_t)(m0 + arow) * Hd + aseg;
  const bf16* bptr = Wt + (size_t)(n0 + bcol) * Hd + bseg;
  const int sAoff = arow * 64 + aseg;
  const int sBoff = bcol * 64 + bseg;

  v8f acc[2][2];
#pragma unroll
  for (int i = 0; i < 2; ++i)
#pragma unroll
    for (int j = 0; j < 2; ++j)
#pragma unroll
      for (int r = 0; r < 8; ++r) acc[i][j][r] = 0.f;

  bf16x8 ra[4], rb[2];

  // prologue: stage 0
#pragma unroll
  for (int j = 0; j < 4; ++j) ra[j] = *(const bf16x8*)(aptr + j * 8);
#pragma unroll
  for (int j = 0; j < 2; ++j) rb[j] = *(const bf16x8*)(bptr + j * 8);
#pragma unroll
  for (int j = 0; j < 4; ++j) *(bf16x8*)&sA[0][sAoff + j * 8] = ra[j];
#pragma unroll
  for (int j = 0; j < 2; ++j) *(bf16x8*)&sB[0][sBoff + j * 8] = rb[j];
  __syncthreads();

  const int NK = Hd / 64;   // 16 stages
#pragma unroll 1
  for (int kt = 0; kt < NK; ++kt) {
    const int cur = kt & 1;
    const bool hasNext = (kt + 1) < NK;
    if (hasNext) {
      const bf16* ap = aptr + (kt + 1) * 64;
      const bf16* bp = bptr + (kt + 1) * 64;
#pragma unroll
      for (int j = 0; j < 4; ++j) ra[j] = *(const bf16x8*)(ap + j * 8);
#pragma unroll
      for (int j = 0; j < 2; ++j) rb[j] = *(const bf16x8*)(bp + j * 8);
      if (kt + 2 < NK) {
        __builtin_prefetch((const void*)(aptr + (kt + 2) * 64), 0, 1);
        __builtin_prefetch((const void*)(bptr + (kt + 2) * 64), 0, 1);
      }
    }
#pragma unroll
    for (int kk = 0; kk < 64; kk += 32) {
      v16bf af[2], bfr[2];
#pragma unroll
      for (int rt = 0; rt < 2; ++rt)
        af[rt] = afrag(&sA[cur][(waveM * 32 + rt * 16) * 64 + kk], 64, mloc, kbA);
#pragma unroll
      for (int ct = 0; ct < 2; ++ct)
        bfr[ct] = bfrag(&sB[cur][(waveN * 32 + ct * 16) * 64 + kk], 64, mloc, kbB);
#pragma unroll
      for (int rt = 0; rt < 2; ++rt)
#pragma unroll
        for (int ct = 0; ct < 2; ++ct)
          acc[rt][ct] = wmma_bf16(af[rt], bfr[ct], acc[rt][ct]);
    }
    if (hasNext) {
#pragma unroll
      for (int j = 0; j < 4; ++j) *(bf16x8*)&sA[cur ^ 1][sAoff + j * 8] = ra[j];
#pragma unroll
      for (int j = 0; j < 2; ++j) *(bf16x8*)&sB[cur ^ 1][sBoff + j * 8] = rb[j];
      __syncthreads();
    }
  }

  // branch-free epilogue
  const int rowOff = (lane < 16) ? 0 : 8;
#pragma unroll
  for (int rt = 0; rt < 2; ++rt)
#pragma unroll
    for (int ct = 0; ct < 2; ++ct) {
      int gcol = n0 + waveN * 32 + ct * 16 + mloc;
      float bv = bias[gcol];
      int growb = m0 + waveM * 32 + rt * 16 + rowOff;
      size_t base = (size_t)growb * Hd + gcol;
#pragma unroll
      for (int r = 0; r < 8; ++r) {
        float v = acc[rt][ct][r] + bv;
        if (F32OUT) outF[base + (size_t)r * Hd] = v;
        else        outB[base + (size_t)r * Hd] = tobf(v);
      }
    }
}

// ---------------- flash cross-attention ----------------
// grid: (ceil(Lq/128), B*NH); 8 waves; each wave owns a 16-row Q tile.
// Q,K bf16 [B*L,1024]; Vt bf16 [B,NH,HD,Lk]; ctx bf16 [B*Lq,1024].
// All tile loads issued up front; softmax denominator accumulated on the matrix
// pipe (P @ ones-column WMMA, same rescale recurrence as O) -- no sum shuffles;
// row-max uses asm-bundled ds_swizzle butterflies (8 swizzles per DS round-trip).

__global__ __launch_bounds__(256) void attn_kernel(
    const bf16* __restrict__ Q, const bf16* __restrict__ K,
    const bf16* __restrict__ Vt, const float* __restrict__ mask,
    bf16* __restrict__ ctx, int Lq, int Lk) {
  __shared__ bf16 ldsP[8][16 * 32];   // per-wave P (16x32) staging, C->A layout

  const int t = threadIdx.x;
  const int lane = t & 31;
  const int wave = t >> 5;
  const int bh = blockIdx.y;
  const int b = bh >> 4;
  const int h = bh & 15;
  const int q0 = blockIdx.x * 128 + wave * 16;
  if (q0 >= Lq) return;

  const int m = lane & 15;
  const int kbA = (lane < 16) ? 0 : 8;
  const int kbB = (lane < 16) ? 0 : 16;
  const int rowOff = (lane < 16) ? 0 : 8;
  bf16* myP = ldsP[wave];

  const bf16* qbase = Q + (size_t)(b * Lq + q0) * Hd + h * HD;
  v16bf aq0 = afrag(qbase, Hd, m, kbA);        // head dims 0..31
  v16bf aq1 = afrag(qbase + 32, Hd, m, kbA);   // head dims 32..63

  v16bf onesB;                                 // 32x16 all-ones B fragment
#pragma unroll
  for (int i = 0; i < 16; ++i) onesB[i] = (bf16)1.0f;

  v8f o[4], oS;                                // oS accumulates row sums of P
#pragma unroll
  for (int n = 0; n < 4; ++n)
#pragma unroll
    for (int r = 0; r < 8; ++r) o[n][r] = 0.f;
#pragma unroll
  for (int r = 0; r < 8; ++r) oS[r] = 0.f;
  float mrow[8];
#pragma unroll
  for (int r = 0; r < 8; ++r) mrow[r] = -1e30f;

  const float scale = 0.125f;   // 1/sqrt(64)

#pragma unroll 1
  for (int kt = 0; kt < Lk; kt += 32) {
    // ---- issue ALL tile loads up front (K + V fragments, mask) ----
    v16bf bk[4];
#pragma unroll
    for (int s = 0; s < 2; ++s) {
      const bf16* kbase = K + (size_t)(b * Lk + kt + s * 16) * Hd + h * HD;
      bk[s * 2 + 0] = bfrag(kbase, Hd, m, kbB);
      bk[s * 2 + 1] = bfrag(kbase + 32, Hd, m, kbB);
    }
    v16bf bv[4];
#pragma unroll
    for (int n = 0; n < 4; ++n) {
      const bf16* vbase = Vt + (((size_t)(b * NH + h)) * HD + n * 16) * Lk + kt;
      bv[n] = bfrag(vbase, Lk, m, kbB);
    }
    float mk0 = mask[b * Lk + kt + m];
    float mk1 = mask[b * Lk + kt + 16 + m];

    // ---- scores: 16x32 tile = two 16x16 WMMA accumulators ----
    v8f sc[2];
#pragma unroll
    for (int s = 0; s < 2; ++s) {
      v8f c;
#pragma unroll
      for (int r = 0; r < 8; ++r) c[r] = 0.f;
      c = wmma_bf16(aq0, bk[s * 2 + 0], c);
      c = wmma_bf16(aq1, bk[s * 2 + 1], c);
      float mk = s ? mk1 : mk0;
#pragma unroll
      for (int r = 0; r < 8; ++r) sc[s][r] = c[r] * scale + mk;
    }

    // ---- row max across the 16 lanes of each half (pipelined butterflies) ----
    float tm[8];
#pragma unroll
    for (int r = 0; r < 8; ++r) tm[r] = fmaxf(sc[0][r], sc[1][r]);
    maxstep8(tm, swz8_x1);
    maxstep8(tm, swz8_x2);
    maxstep8(tm, swz8_x4);
    maxstep8(tm, swz8_x8);

    float al[8];
#pragma unroll
    for (int r = 0; r < 8; ++r) {
      float nm = fmaxf(mrow[r], tm[r]);
      al[r] = __expf(mrow[r] - nm);
      mrow[r] = nm;
      sc[0][r] = __expf(sc[0][r] - nm);
      sc[1][r] = __expf(sc[1][r] - nm);
    }
#pragma unroll
    for (int r = 0; r < 8; ++r) oS[r] *= al[r];
#pragma unroll
    for (int n = 0; n < 4; ++n)
#pragma unroll
      for (int r = 0; r < 8; ++r) o[n][r] *= al[r];

    // ---- P: C-layout -> LDS -> A-layout fragment (same wave, DS in-order) ----
#pragma unroll
    for (int r = 0; r < 8; ++r) {
      myP[(r + rowOff) * 32 + m]      = tobf(sc[0][r]);
      myP[(r + rowOff) * 32 + 16 + m] = tobf(sc[1][r]);
    }
    v16bf ap = afrag(myP, 32, m, kbA);
#pragma unroll
    for (int n = 0; n < 4; ++n) o[n] = wmma_bf16(ap, bv[n], o[n]);
    oS = wmma_bf16(ap, onesB, oS);   // denominator on the matrix pipe
  }

#pragma unroll
  for (int n = 0; n < 4; ++n)
#pragma unroll
    for (int r = 0; r < 8; ++r) {
      float inv = 1.f / oS[r];
      size_t idx = (size_t)(b * Lq + q0 + r + rowOff) * Hd + h * HD + n * 16 + m;
      ctx[idx] = tobf(o[n][r] * inv);
    }
}

// ---------------- residual + LayerNorm ----------------

__global__ __launch_bounds__(256) void ln_kernel(const float* __restrict__ Y,
                                                 const float* __restrict__ res,
                                                 const float* __restrict__ g,
                                                 const float* __restrict__ beta,
                                                 float* __restrict__ out) {
  const int row = blockIdx.x;
  const int t = threadIdx.x;
  const float* y = Y + (size_t)row * Hd;
  const float* r = res + (size_t)row * Hd;
  float x[4], s = 0.f, ss = 0.f;
#pragma unroll
  for (int i = 0; i < 4; ++i) {
    int c = t + i * 256;
    x[i] = y[c] + r[c];
    s += x[i]; ss += x[i] * x[i];
  }
  // wave32 butterfly via ds_swizzle (xor 1..16 within group of 32)
  s += swz<1>(s);  ss += swz<1>(ss);
  s += swz<2>(s);  ss += swz<2>(ss);
  s += swz<4>(s);  ss += swz<4>(ss);
  s += swz<8>(s);  ss += swz<8>(ss);
  s += swz<16>(s); ss += swz<16>(ss);
  __shared__ float shs[8], shss[8];
  int wave = t >> 5, lane = t & 31;
  if (lane == 0) { shs[wave] = s; shss[wave] = ss; }
  __syncthreads();
  if (t == 0) {
    float S = 0.f, SS = 0.f;
    for (int w = 0; w < 8; ++w) { S += shs[w]; SS += shss[w]; }
    shs[0] = S; shss[0] = SS;
  }
  __syncthreads();
  float mu = shs[0] * (1.f / 1024.f);
  float var = shss[0] * (1.f / 1024.f) - mu * mu;
  float inv = rsqrtf(var + 1e-12f);
  float* o = out + (size_t)row * Hd;
#pragma unroll
  for (int i = 0; i < 4; ++i) {
    int c = t + i * 256;
    o[c] = (x[i] - mu) * inv * g[c] + beta[c];
  }
}

// ---------------- host orchestration ----------------

extern "C" void kernel_launch(void* const* d_in, const int* in_sizes, int n_in,
                              void* d_out, int out_size, void* d_ws, size_t ws_size,
                              hipStream_t stream) {
  (void)in_sizes; (void)n_in; (void)out_size; (void)ws_size;

  const float* visH = (const float*)d_in[0];
  const float* txtH = (const float*)d_in[1];
  const float* visM = (const float*)d_in[2];
  const float* txtM = (const float*)d_in[3];
  // params: w/b pairs: 0 v2t_q, 1 v2t_k, 2 v2t_v, 3 t2v_q, 4 t2v_k, 5 t2v_v,
  //                    6 vision_out, 7 text_out; then ln g/b pairs.
  const float* Wf[8];
  const float* Bf[8];
  for (int i = 0; i < 8; ++i) {
    Wf[i] = (const float*)d_in[4 + 2 * i];
    Bf[i] = (const float*)d_in[5 + 2 * i];
  }
  const float* vln_g = (const float*)d_in[20];
  const float* vln_b = (const float*)d_in[21];
  const float* tln_g = (const float*)d_in[22];
  const float* tln_b = (const float*)d_in[23];

  const size_t MV = (size_t)Bb * LV;   // 4608
  const size_t MT = (size_t)Bb * LT;   // 4096

  char* ws = (char*)d_ws;
  size_t off = 0;
  auto alloc = [&](size_t bytes) -> void* {
    void* p = ws + off;
    off = (off + bytes + 255) & ~(size_t)255;
    return p;
  };
  bf16* Wt[8];
  for (int i = 0; i < 8; ++i) Wt[i] = (bf16*)alloc((size_t)Hd * Hd * 2);
  bf16* actV = (bf16*)alloc(MV * Hd * 2);
  bf16* actT = (bf16*)alloc(MT * Hd * 2);
  bf16* Qv   = (bf16*)alloc(MV * Hd * 2);
  bf16* Kt   = (bf16*)alloc(MT * Hd * 2);
  bf16* VtmpT= (bf16*)alloc(MT * Hd * 2);
  bf16* VtT  = (bf16*)alloc(MT * Hd * 2);
  bf16* Qt   = (bf16*)alloc(MT * Hd * 2);
  bf16* Kv   = (bf16*)alloc(MV * Hd * 2);
  bf16* VtmpV= (bf16*)alloc(MV * Hd * 2);
  bf16* VtV  = (bf16*)alloc(MV * Hd * 2);
  bf16* ctxV = (bf16*)alloc(MV * Hd * 2);
  bf16* ctxT = (bf16*)alloc(MT * Hd * 2);
  float* Yv  = (float*)alloc(MV * Hd * 4);
  float* Yt  = (float*)alloc(MT * Hd * 4);

  // 1) weight convert + transpose (bf16, [N][K])
  for (int i = 0; i < 8; ++i)
    cvt_wT_kernel<<<(Hd * Hd) / 256, 256, 0, stream>>>(Wf[i], Wt[i]);

  // 2) activation convert
  cvt_bf16_kernel<<<(int)((MV * Hd) / 256), 256, 0, stream>>>(visH, actV, (int)(MV * Hd));
  cvt_bf16_kernel<<<(int)((MT * Hd) / 256), 256, 0, stream>>>(txtH, actT, (int)(MT * Hd));

  // 3) QKV projections (bf16 outputs)
  dim3 gv((unsigned)(MV / 128), Hd / 64), gt((unsigned)(MT / 128), Hd / 64);
  gemm_bf16_kernel<false><<<gv, 256, 0, stream>>>(actV, Wt[0], Bf[0], nullptr, Qv, (int)MV);
  gemm_bf16_kernel<false><<<gt, 256, 0, stream>>>(actT, Wt[4], Bf[4], nullptr, Kt, (int)MT);
  gemm_bf16_kernel<false><<<gt, 256, 0, stream>>>(actT, Wt[5], Bf[5], nullptr, VtmpT, (int)MT);
  gemm_bf16_kernel<false><<<gt, 256, 0, stream>>>(actT, Wt[3], Bf[3], nullptr, Qt, (int)MT);
  gemm_bf16_kernel<false><<<gv, 256, 0, stream>>>(actV, Wt[1], Bf[1], nullptr, Kv, (int)MV);
  gemm_bf16_kernel<false><<<gv, 256, 0, stream>>>(actV, Wt[2], Bf[2], nullptr, VtmpV, (int)MV);

  // 4) V transposes for PV B-fragments
  vtrans_kernel<<<(int)((MT * Hd) / 256), 256, 0, stream>>>(VtmpT, VtT, LT);
  vtrans_kernel<<<(int)((MV * Hd) / 256), 256, 0, stream>>>(VtmpV, VtV, LV);

  // 5) cross attentions (flash, WMMA)
  dim3 av((LV + 127) / 128, Bb * NH), at((LT + 127) / 128, Bb * NH);
  attn_kernel<<<av, 256, 0, stream>>>(Qv, Kt, VtT, txtM, ctxV, LV, LT);
  attn_kernel<<<at, 256, 0, stream>>>(Qt, Kv, VtV, visM, ctxT, LT, LV);

  // 6) output projections (f32)
  gemm_bf16_kernel<true><<<gv, 256, 0, stream>>>(ctxV, Wt[6], Bf[6], Yv, nullptr, (int)MV);
  gemm_bf16_kernel<true><<<gt, 256, 0, stream>>>(ctxT, Wt[7], Bf[7], Yt, nullptr, (int)MT);

  // 7) residual + LayerNorm into d_out (vision first, then text)
  float* outV = (float*)d_out;
  float* outT = outV + MV * Hd;
  ln_kernel<<<(unsigned)MV, 256, 0, stream>>>(Yv, visH, vln_g, vln_b, outV);
  ln_kernel<<<(unsigned)MT, 256, 0, stream>>>(Yt, txtH, tln_g, tln_b, outT);
}
